// TriPathDecoderBlock_40939628266032
// MI455X (gfx1250) — matrix operand order repaired and checked by reading, not compiled, not run on hardware
//
#include <hip/hip_runtime.h>

typedef __attribute__((ext_vector_type(16))) _Float16 v16h;
typedef __attribute__((ext_vector_type(8)))  _Float16 v8h;
typedef __attribute__((ext_vector_type(8)))  float    v8f;
typedef __attribute__((ext_vector_type(4)))  float    v4f;

namespace {

constexpr int BB    = 4;
constexpr int CIN   = 256;
constexpr int CD    = 128;
constexpr int H0    = 64,  W0C = 64;
constexpr int H1    = 128, W1C = 128;
constexpr int N0    = H0 * W0C;   // 4096
constexpr int N1    = H1 * W1C;   // 16384
constexpr int NHEAD = 4;
constexpr int HD    = 32;

// ---------------------------------------------------------------------------
// WMMA core (CDNA5 v_wmma_f32_16x16x32_f16, fragments use contiguous k order, k = 16*(lane>=16) + i).
// All activations are f16; GEMM B inputs are pixel-major [b][pix][C] and
// weights are f16 with K contiguous, so A and B fragments are DIRECT global
// vector loads -- no LDS, no barriers, no staging in any matrix kernel.
//
// A frag (lane): rows m=lane&15; halves i = K koffA+i,
//                koffA = 16*(lane>=16)
// B frag (lane): col n=lane&15; halves i = K koffB+i, koffB = 16*(lane>=16)
// ---------------------------------------------------------------------------
__device__ __forceinline__ v8f wmma_ff16(v16h a, v16h b, v8f c) {
  return __builtin_amdgcn_wmma_f32_16x16x32_f16(false, a, false, b, (short)0, c,
                                                false, false);
}

__device__ __forceinline__ v16h cat8(v8h lo, v8h hi) {
  return __builtin_shufflevector(lo, hi, 0, 1, 2, 3, 4, 5, 6, 7, 8, 9, 10, 11,
                                 12, 13, 14, 15);
}

__device__ __forceinline__ v16h fragA_f16(const _Float16* row, int koffA) {
  return cat8(*(const v8h*)&row[koffA], *(const v8h*)&row[8 + koffA]);
}

__device__ __forceinline__ v8h cvt8(v8f a) {
  v8h r;
#pragma unroll
  for (int i = 0; i < 8; ++i) r[i] = (_Float16)a[i];
  return r;
}

// C/D 16x16 f32: lane col n = lane&15, rows m0..m0+7, m0 = 8*(lane>=16)
__device__ __forceinline__ void store_C(float* C, int ldC, v8f acc) {
  const int lane = threadIdx.x & 31;
  const int n    = lane & 15;
  const int m0   = (lane >> 4) << 3;
#pragma unroll
  for (int r = 0; r < 8; ++r) C[(size_t)(m0 + r) * ldC + n] = acc[r];
}

// ---------------------------------------------------------------------------
// Implicit-GEMM 3x3 conv (pad=1). in: f16 pixel-major [b][pix][Cin];
// wt: f16 tap-major [9][Cout][Cin]; out: f32 channel-major.
// grid=(Cout/16, N/64, B), block=128 (4 waves). Branch-free borders via
// clamped addresses + selects (tiles are row-aligned since W%16==0).
// ---------------------------------------------------------------------------
__global__ void __launch_bounds__(128)
conv3x3_wmma_kernel(const _Float16* __restrict__ in,
                    const _Float16* __restrict__ wt, float* __restrict__ out,
                    int Cin, int Cout, int H, int W, int CtotOut, int coOff) {
  const int N      = H * W;
  const int coTile = blockIdx.x * 16;
  const int wave   = threadIdx.x >> 5;
  const int lane   = threadIdx.x & 31;
  const int b      = blockIdx.z;
  const int myPix  = blockIdx.y * 64 + wave * 16;

  const int m     = lane & 15;          // A row / B pixel within tile
  const int koffA = (lane >> 4) << 4;
  const int koffB = (lane >> 4) << 4;
  const int yRow  = myPix / W;          // uniform: tile lies in one row
  const int x0    = myPix % W;

  const _Float16* inB = in + (size_t)b * N * Cin;
  const v16h zf = {};
  v8f acc = {};

  for (int t = 0; t < 9; ++t) {
    const int dy = t / 3 - 1;
    const int dx = t % 3 - 1;
    const int y  = yRow + dy;
    const bool yok = (unsigned)y < (unsigned)H;
    const int yc = yok ? y : 0;
    const int x  = x0 + m + dx;
    const bool ok = yok && ((unsigned)x < (unsigned)W);
    const int xc = x < 0 ? 0 : (x >= W ? W - 1 : x);
    const _Float16* prow = inB + ((size_t)yc * W + xc) * Cin;
    const _Float16* wrow = wt + (size_t)t * Cout * Cin + (size_t)(coTile + m) * Cin;
    for (int c0 = 0; c0 < Cin; c0 += 32) {
      __builtin_prefetch(&prow[c0 + 256], 0, 1);
      const v16h fa = fragA_f16(&wrow[c0], koffA);
      v16h fb = *(const v16h*)&prow[c0 + koffB];
      fb = ok ? fb : zf;
      acc = wmma_ff16(fa, fb, acc);
    }
  }
  float* o = out + ((size_t)b * CtotOut + coOff + coTile) * N + myPix;
  store_C(o, N, acc);
}

// ---------------------------------------------------------------------------
// 1x1 channel-mixing GEMM. in: f16 pixel-major [b][pix][Cin]; wf: f16
// [Cout][Cin]. Outputs (each optional): f32 channel-major, f16 channel-major
// (attention q/k), f16 pixel-major (vectorized 16B stores; attention v).
// grid=(Cout/16, N/64, B), block=128.
// ---------------------------------------------------------------------------
__global__ void __launch_bounds__(128)
gemm1x1_wmma_kernel(const _Float16* __restrict__ in,
                    const _Float16* __restrict__ wf,
                    const float* __restrict__ bias, float* __restrict__ out,
                    _Float16* __restrict__ outh, _Float16* __restrict__ outT,
                    int Cin, int Cout, int N) {
  const int coTile = blockIdx.x * 16;
  const int wave   = threadIdx.x >> 5;
  const int lane   = threadIdx.x & 31;
  const int b      = blockIdx.z;
  const int myPix  = blockIdx.y * 64 + wave * 16;

  const int m     = lane & 15;
  const int koffA = (lane >> 4) << 4;
  const int koffB = (lane >> 4) << 4;

  const _Float16* inP = in + ((size_t)b * N + myPix + m) * Cin;
  const _Float16* wr  = wf + (size_t)(coTile + m) * Cin;

  v8f acc = {};
  for (int c0 = 0; c0 < Cin; c0 += 32) {
    __builtin_prefetch(&inP[c0 + 256], 0, 1);
    const v16h fa = fragA_f16(&wr[c0], koffA);
    const v16h fb = *(const v16h*)&inP[c0 + koffB];
    acc = wmma_ff16(fa, fb, acc);
  }
  const int n  = lane & 15;
  const int m0 = (lane >> 4) << 3;
  v8f vals;
#pragma unroll
  for (int r = 0; r < 8; ++r)
    vals[r] = acc[r] + (bias ? bias[coTile + m0 + r] : 0.0f);
  if (out) {
#pragma unroll
    for (int r = 0; r < 8; ++r)
      out[((size_t)b * Cout + coTile + m0 + r) * N + myPix + n] = vals[r];
  }
  if (outh) {
#pragma unroll
    for (int r = 0; r < 8; ++r)
      outh[((size_t)b * Cout + coTile + m0 + r) * N + myPix + n] =
          (_Float16)vals[r];
  }
  if (outT) {  // pixel-major: 8 contiguous channels -> one 16B store
    *(v8h*)&outT[((size_t)b * N + myPix + n) * Cout + coTile + m0] = cvt8(vals);
  }
}

// ---------------------------------------------------------------------------
// Attention scores + softmax per (p,b,h): S = (q k^T)/sqrt(hd), N=16384.
// q/k channel-major f16: the GEMM K-dim (pixels) is contiguous, so all
// fragments are direct global vector loads. grid=3*B*NHEAD, block=128.
// ---------------------------------------------------------------------------
__global__ void __launch_bounds__(128)
attn_scores_kernel(const _Float16* __restrict__ q,
                   const _Float16* __restrict__ k,
                   _Float16* __restrict__ attnP, int N) {
  const int pbh  = blockIdx.x;
  const int h    = pbh % NHEAD;
  const int bt   = pbh / NHEAD;
  const int b    = bt % BB;
  const int p    = bt / BB;
  const int wave = threadIdx.x >> 5;
  const int lane = threadIdx.x & 31;
  const int tid  = threadIdx.x;

  __shared__ float Sp[4][32 * 32];
  __shared__ float Sf[32 * 32];

  const _Float16* qp = q + (((size_t)p * BB + b) * CD + h * HD) * N;
  const _Float16* kp = k + (((size_t)p * BB + b) * CD + h * HD) * N;

  const int m     = lane & 15;
  const int koffA = (lane >> 4) << 4;
  const int koffB = (lane >> 4) << 4;

  v8f a00 = {}, a01 = {}, a10 = {}, a11 = {};
  const int slice = N / 4;
  for (int k0 = wave * slice; k0 < (wave + 1) * slice; k0 += 32) {
    __builtin_prefetch(&qp[(size_t)m * N + k0 + 256], 0, 1);
    const v16h fa0 = cat8(*(const v8h*)&qp[(size_t)m * N + k0 + koffA],
                          *(const v8h*)&qp[(size_t)m * N + k0 + 8 + koffA]);
    const v16h fa1 =
        cat8(*(const v8h*)&qp[(size_t)(16 + m) * N + k0 + koffA],
             *(const v8h*)&qp[(size_t)(16 + m) * N + k0 + 8 + koffA]);
    const v16h fb0 = *(const v16h*)&kp[(size_t)m * N + k0 + koffB];
    const v16h fb1 = *(const v16h*)&kp[(size_t)(16 + m) * N + k0 + koffB];
    a00 = wmma_ff16(fa0, fb0, a00);
    a01 = wmma_ff16(fa0, fb1, a01);
    a10 = wmma_ff16(fa1, fb0, a10);
    a11 = wmma_ff16(fa1, fb1, a11);
  }
  store_C(&Sp[wave][0], 32, a00);
  store_C(&Sp[wave][16], 32, a01);
  store_C(&Sp[wave][16 * 32], 32, a10);
  store_C(&Sp[wave][16 * 32 + 16], 32, a11);
  __syncthreads();
  const float sc = 0.17677669529663687f;  // 1/sqrt(32)
  for (int i = tid; i < 1024; i += 128)
    Sf[i] = (Sp[0][i] + Sp[1][i] + Sp[2][i] + Sp[3][i]) * sc;
  __syncthreads();
  if (tid < 32) {  // softmax over e, one row per lane
    const int d = tid;
    float mx = -3.0e38f;
#pragma unroll
    for (int e = 0; e < 32; ++e) mx = fmaxf(mx, Sf[d * 32 + e]);
    float ex[32];
    float s = 0.0f;
#pragma unroll
    for (int e = 0; e < 32; ++e) {
      ex[e] = expf(Sf[d * 32 + e] - mx);
      s += ex[e];
    }
    const float inv = 1.0f / s;
    _Float16* o = attnP + (size_t)pbh * 1024 + d * 32;
#pragma unroll
    for (int e = 0; e < 32; ++e) o[e] = (_Float16)(ex[e] * inv);
  }
}

// ---------------------------------------------------------------------------
// out = attn @ v. Fully LDS-free: A (probs) and B (pixel-major v) fragments
// are direct global vector loads; result written into pixel-major tri with
// contiguous 16B stores. grid=(N/64, 3*B*NHEAD), block=128.
// ---------------------------------------------------------------------------
__global__ void __launch_bounds__(128)
attn_apply_kernel(const _Float16* __restrict__ attnP,
                  const _Float16* __restrict__ vT, _Float16* __restrict__ triT,
                  int N) {
  const int pbh  = blockIdx.y;
  const int h    = pbh % NHEAD;
  const int bt   = pbh / NHEAD;
  const int b    = bt % BB;
  const int p    = bt / BB;
  const int wave = threadIdx.x >> 5;
  const int lane = threadIdx.x & 31;

  const _Float16* ap  = attnP + (size_t)pbh * 1024;
  const _Float16* vpT = vT + ((size_t)p * BB + b) * (size_t)N * CD;
  const int myPix     = blockIdx.x * 64 + wave * 16;

  const int m     = lane & 15;
  const int koffA = (lane >> 4) << 4;
  const int koffB = (lane >> 4) << 4;

  const v16h fa0 = cat8(*(const v8h*)&ap[(size_t)m * 32 + koffA],
                        *(const v8h*)&ap[(size_t)m * 32 + 8 + koffA]);
  const v16h fa1 = cat8(*(const v8h*)&ap[(size_t)(16 + m) * 32 + koffA],
                        *(const v8h*)&ap[(size_t)(16 + m) * 32 + 8 + koffA]);
  const v16h fb =
      *(const v16h*)&vpT[(size_t)(myPix + m) * CD + h * HD + koffB];
  v8f acc0 = {}, acc1 = {};
  acc0 = wmma_ff16(fa0, fb, acc0);
  acc1 = wmma_ff16(fa1, fb, acc1);
  const int m0 = (lane >> 4) << 3;
  _Float16* o =
      triT + ((size_t)b * N + myPix + (lane & 15)) * 384 + p * 128 + h * HD;
  *(v8h*)&o[m0]      = cvt8(acc0);
  *(v8h*)&o[16 + m0] = cvt8(acc1);
}

// ---------------------------------------------------------------------------
// Weight prep: f32 -> f16 (flat), and conv weights to tap-major [9][O][I]
// ---------------------------------------------------------------------------
__global__ void wcvt_kernel(const float* __restrict__ in,
                            _Float16* __restrict__ out, size_t total) {
  const size_t i = (size_t)blockIdx.x * blockDim.x + threadIdx.x;
  if (i < total) out[i] = (_Float16)in[i];
}

__global__ void wprep_conv_kernel(const float* __restrict__ w,
                                  _Float16* __restrict__ o, int Cout,
                                  int Cin) {
  const size_t oc = (size_t)Cout * Cin;
  const size_t total = oc * 9;
  const size_t i = (size_t)blockIdx.x * blockDim.x + threadIdx.x;
  if (i >= total) return;
  const size_t t = i / oc;
  const size_t r = i % oc;
  o[i] = (_Float16)w[r * 9 + t];
}

// activations f32 [b][c][n] -> f16 pixel-major [b][n][c], 4 pixels/thread
__global__ void cvt_pm_kernel(const float* __restrict__ in,
                              _Float16* __restrict__ out, int B, int C,
                              int N) {
  const size_t total4 = (size_t)B * C * (N / 4);
  const size_t i = (size_t)blockIdx.x * blockDim.x + threadIdx.x;
  if (i >= total4) return;
  const int n4 = (int)(i % (N / 4)) * 4;
  size_t t     = i / (N / 4);
  const int c  = (int)(t % C);
  const int b  = (int)(t / C);
  const v4f v  = *(const v4f*)&in[((size_t)b * C + c) * N + n4];
#pragma unroll
  for (int u = 0; u < 4; ++u)
    out[((size_t)b * N + n4 + u) * C + c] = (_Float16)v[u];
}

// align-corners bilinear resize, f32 [b][c][hw] -> f16 pixel-major [b][hw][c]
__global__ void resize_pm_kernel(const float* __restrict__ in,
                                 _Float16* __restrict__ out, int B, int C,
                                 int Hin, int Win, int Hout, int Wout) {
  const size_t total = (size_t)B * C * Hout * Wout;
  size_t idx = (size_t)blockIdx.x * blockDim.x + threadIdx.x;
  if (idx >= total) return;
  const int j = (int)(idx % Wout);
  size_t t    = idx / Wout;
  const int i = (int)(t % Hout);
  t /= Hout;
  const int c = (int)(t % C);
  const int b = (int)(t / C);
  const float ys = (float)i * (float)(Hin - 1) / (float)(Hout - 1);
  const float xs = (float)j * (float)(Win - 1) / (float)(Wout - 1);
  int y0 = (int)floorf(ys);
  int x0 = (int)floorf(xs);
  int y1 = y0 + 1; if (y1 > Hin - 1) y1 = Hin - 1;
  int x1 = x0 + 1; if (x1 > Win - 1) x1 = Win - 1;
  const float wy = ys - (float)y0;
  const float wx = xs - (float)x0;
  const float* p = in + ((size_t)b * C + c) * Hin * Win;
  const float r0 = p[y0 * Win + x0] * (1.f - wx) + p[y0 * Win + x1] * wx;
  const float r1 = p[y1 * Win + x0] * (1.f - wx) + p[y1 * Win + x1] * wx;
  const float v  = r0 * (1.f - wy) + r1 * wy;
  out[((size_t)b * Hout * Wout + (size_t)i * Wout + j) * C + c] = (_Float16)v;
}

// Haar IDWT: f32 ch-major [ll|lh|hl|hh] (B,128,64,64) -> f16 pixel-major rec
__global__ void idwt_kernel(const float* __restrict__ a,
                            _Float16* __restrict__ rec) {
  const size_t total = (size_t)BB * 32 * N0;
  size_t i = (size_t)blockIdx.x * blockDim.x + threadIdx.x;
  if (i >= total) return;
  const int x = (int)(i % W0C);
  size_t t    = i / W0C;
  const int y = (int)(t % H0);
  t /= H0;
  const int c = (int)(t % 32);
  const int b = (int)(t / 32);
  const size_t base = ((size_t)b * 128 + c) * N0 + (size_t)y * W0C + x;
  const float ll = a[base];
  const float lh = a[base + (size_t)32 * N0];
  const float hl = a[base + (size_t)64 * N0];
  const float hh = a[base + (size_t)96 * N0];
  const int W2 = 2 * W0C;
  _Float16* o = rec + ((size_t)b * 4 * N0) * 32 + c;
  o[((size_t)(2 * y) * W2 + 2 * x) * 32]     = (_Float16)(0.5f * (ll + lh + hl + hh));
  o[((size_t)(2 * y) * W2 + 2 * x + 1) * 32] = (_Float16)(0.5f * (ll - lh + hl - hh));
  o[((size_t)(2 * y + 1) * W2 + 2 * x) * 32] = (_Float16)(0.5f * (ll + lh - hl - hh));
  o[((size_t)(2 * y + 1) * W2 + 2 * x + 1) * 32] =
      (_Float16)(0.5f * (ll - lh - hl + hh));
}

// ---------------------------------------------------------------------------
// GroupNorm / BatchNorm stats + apply, 4-wide vectorized. Apply epilogues:
// act 0=none, 1=relu, 2=tanh*clip(scale,1,5), 3=silu; optional pre-add,
// optional f32 out (ch-major) and f16 twin out (pixel-major, next WMMA input)
// ---------------------------------------------------------------------------
__global__ void __launch_bounds__(256)
gn_stats_kernel(const float* __restrict__ in, const float* __restrict__ pre,
                float* __restrict__ stats, int Ctot, int c0, int C, int G,
                int N) {
  const int b   = blockIdx.x / G;
  const int g   = blockIdx.x % G;
  const int cpg = C / G;
  const int N4  = N / 4;
  const size_t cnt4 = (size_t)cpg * N4;
  float s = 0.f, ss = 0.f;
  for (size_t i = threadIdx.x; i < cnt4; i += 256) {
    const int cc = (int)(i / N4);
    const int n4 = (int)(i % N4) * 4;
    const size_t idx = ((size_t)b * Ctot + c0 + g * cpg + cc) * N + n4;
    v4f v = *(const v4f*)&in[idx];
    if (pre) v += *(const v4f*)&pre[idx];
#pragma unroll
    for (int u = 0; u < 4; ++u) {
      s += v[u];
      ss += v[u] * v[u];
    }
  }
  __shared__ float red[512];
  red[threadIdx.x]       = s;
  red[256 + threadIdx.x] = ss;
  __syncthreads();
  for (int st = 128; st > 0; st >>= 1) {
    if ((int)threadIdx.x < st) {
      red[threadIdx.x] += red[threadIdx.x + st];
      red[256 + threadIdx.x] += red[256 + threadIdx.x + st];
    }
    __syncthreads();
  }
  if (threadIdx.x == 0) {
    const float cnt = (float)((size_t)cpg * N);
    const float mn  = red[0] / cnt;
    const float var = red[256] / cnt - mn * mn;
    stats[blockIdx.x * 2 + 0] = mn;
    stats[blockIdx.x * 2 + 1] = rsqrtf(var + 1e-5f);
  }
}

__global__ void gn_apply_kernel(const float* __restrict__ in,
                                const float* __restrict__ pre,
                                float* __restrict__ out,
                                _Float16* __restrict__ outh,  // pixel-major
                                const float* __restrict__ gamma,
                                const float* __restrict__ beta,
                                const float* __restrict__ stats,
                                const float* __restrict__ scl, int act, int B,
                                int Ctot, int c0, int C, int G, int N) {
  const int N4 = N / 4;
  const size_t total4 = (size_t)B * C * N4;
  const size_t i = (size_t)blockIdx.x * blockDim.x + threadIdx.x;
  if (i >= total4) return;
  const int n4 = (int)(i % N4) * 4;
  size_t t     = i / N4;
  const int c  = (int)(t % C);
  const int b  = (int)(t / C);
  const int g  = c / (C / G);
  const size_t idx = ((size_t)b * Ctot + c0 + c) * N + n4;
  v4f v = *(const v4f*)&in[idx];
  if (pre) v += *(const v4f*)&pre[idx];
  const float mn = stats[(b * G + g) * 2 + 0];
  const float rs = stats[(b * G + g) * 2 + 1];
  const float ga = gamma[c], be = beta[c];
  const float sC = (act == 2) ? fminf(fmaxf(scl[0], 1.f), 5.f) : 0.f;
#pragma unroll
  for (int u = 0; u < 4; ++u) {
    float x = (v[u] - mn) * rs * ga + be;
    if (act == 1) x = fmaxf(x, 0.f);
    else if (act == 2) x = tanhf(x) * sC;
    else if (act == 3) x = x / (1.f + expf(-x));
    v[u] = x;
  }
  if (out) *(v4f*)&out[idx] = v;
  if (outh) {
#pragma unroll
    for (int u = 0; u < 4; ++u)
      outh[((size_t)b * N + n4 + u) * C + c] = (_Float16)v[u];
  }
}

__global__ void __launch_bounds__(256)
bn_stats_kernel(const float* __restrict__ in, float* __restrict__ stats, int B,
                int C, int N) {
  const int c  = blockIdx.x;
  const int N4 = N / 4;
  const size_t cnt4 = (size_t)B * N4;
  float s = 0.f, ss = 0.f;
  for (size_t i = threadIdx.x; i < cnt4; i += 256) {
    const int b  = (int)(i / N4);
    const int n4 = (int)(i % N4) * 4;
    const v4f v  = *(const v4f*)&in[((size_t)b * C + c) * N + n4];
#pragma unroll
    for (int u = 0; u < 4; ++u) {
      s += v[u];
      ss += v[u] * v[u];
    }
  }
  __shared__ float red[512];
  red[threadIdx.x]       = s;
  red[256 + threadIdx.x] = ss;
  __syncthreads();
  for (int st = 128; st > 0; st >>= 1) {
    if ((int)threadIdx.x < st) {
      red[threadIdx.x] += red[threadIdx.x + st];
      red[256 + threadIdx.x] += red[256 + threadIdx.x + st];
    }
    __syncthreads();
  }
  if (threadIdx.x == 0) {
    const float cnt = (float)((size_t)B * N);
    const float mn  = red[0] / cnt;
    const float var = red[256] / cnt - mn * mn;
    stats[c * 2 + 0] = mn;
    stats[c * 2 + 1] = rsqrtf(var + 1e-5f);
  }
}

// out = relu(bn(in)) + post(optional); f32 out / f16 pixel-major twin
__global__ void bn_apply_kernel(const float* __restrict__ in,
                                const float* __restrict__ post,
                                float* __restrict__ out,
                                _Float16* __restrict__ outh,
                                const float* __restrict__ gamma,
                                const float* __restrict__ beta,
                                const float* __restrict__ stats, int B, int C,
                                int N) {
  const int N4 = N / 4;
  const size_t total4 = (size_t)B * C * N4;
  const size_t i = (size_t)blockIdx.x * blockDim.x + threadIdx.x;
  if (i >= total4) return;
  const int n4 = (int)(i % N4) * 4;
  size_t t     = i / N4;
  const int c  = (int)(t % C);
  const int b  = (int)(t / C);
  const size_t idx = ((size_t)b * C + c) * N + n4;
  const float mn = stats[c * 2 + 0], rs = stats[c * 2 + 1];
  const float ga = gamma[c], be = beta[c];
  v4f v = *(const v4f*)&in[idx];
#pragma unroll
  for (int u = 0; u < 4; ++u) v[u] = fmaxf((v[u] - mn) * rs * ga + be, 0.f);
  if (post) v += *(const v4f*)&post[idx];
  if (out) *(v4f*)&out[idx] = v;
  if (outh) {
#pragma unroll
    for (int u = 0; u < 4; ++u)
      outh[((size_t)b * N + n4 + u) * C + c] = (_Float16)v[u];
  }
}

}  // namespace

// ---------------------------------------------------------------------------
extern "C" void kernel_launch(void* const* d_in, const int* in_sizes, int n_in,
                              void* d_out, int out_size, void* d_ws,
                              size_t ws_size, hipStream_t stream) {
  (void)in_sizes; (void)n_in; (void)out_size; (void)ws_size;
  const float* X     = (const float*)d_in[0];
  const float* SD    = (const float*)d_in[1];
  const float* SS    = (const float*)d_in[2];   // already Ho x Wo
  const float* SG    = (const float*)d_in[3];
  const float* Wup   = (const float*)d_in[4];
  const float* bnupg = (const float*)d_in[5];
  const float* bnupb = (const float*)d_in[6];
  const float* Wll   = (const float*)d_in[7];
  const float* gllg  = (const float*)d_in[8];
  const float* gllb  = (const float*)d_in[9];
  const float* Whi   = (const float*)d_in[10];
  const float* ghig  = (const float*)d_in[11];
  const float* ghib  = (const float*)d_in[12];
  const float* wscl  = (const float*)d_in[13];
  const float* Wadj  = (const float*)d_in[14];
  const float* gadjg = (const float*)d_in[15];
  const float* gadjb = (const float*)d_in[16];
  const float* Wq    = (const float*)d_in[17];
  const float* bq    = (const float*)d_in[18];
  const float* Wk    = (const float*)d_in[19];
  const float* bk    = (const float*)d_in[20];
  const float* Wv    = (const float*)d_in[21];
  const float* bv    = (const float*)d_in[22];
  const float* Wfu   = (const float*)d_in[23];
  const float* bfu   = (const float*)d_in[24];
  const float* gfg   = (const float*)d_in[25];
  const float* gfb   = (const float*)d_in[26];
  const float* gog   = (const float*)d_in[27];
  const float* gob   = (const float*)d_in[28];
  const float* Wr1   = (const float*)d_in[29];
  const float* bn1g  = (const float*)d_in[30];
  const float* bn1b  = (const float*)d_in[31];
  const float* Wr2   = (const float*)d_in[32];
  const float* bn2g  = (const float*)d_in[33];
  const float* bn2b  = (const float*)d_in[34];

  char* base = (char*)d_ws;
  size_t off = 0;
  auto allocF = [&](size_t n) {
    off = (off + 63) & ~(size_t)63;
    float* p = (float*)(base + off);
    off += n * sizeof(float);
    return p;
  };
  auto allocH = [&](size_t n) {
    off = (off + 63) & ~(size_t)63;
    _Float16* p = (_Float16*)(base + off);
    off += n * sizeof(_Float16);
    return p;
  };

  // f32 (norm inputs / residuals, channel-major)
  float* bufA  = allocF((size_t)BB * 128 * N0);  // detail conv out ll|lh|hl|hh
  float* skipU = allocF((size_t)BB * CD * N1);
  float* mrgd  = allocF((size_t)BB * CD * N1);
  float* fused = allocF((size_t)BB * CD * N1);   // reused for conv r2 out
  float* r1b   = allocF((size_t)BB * CD * N1);
  float* stats = allocF(4096);
  // f16 weights (K contiguous; conv weights tap-major [9][O][I])
  _Float16* Wll16 = allocH((size_t)9 * 32 * CD);
  _Float16* Whi16 = allocH((size_t)9 * 96 * CD);
  _Float16* Wup16 = allocH((size_t)9 * CD * CIN);
  _Float16* Wr116 = allocH((size_t)9 * CD * CD);
  _Float16* Wr216 = allocH((size_t)9 * CD * CD);
  _Float16* Wadj16 = allocH((size_t)CD * 32);
  _Float16* Wq16  = allocH((size_t)3 * CD * CD);
  _Float16* Wk16  = allocH((size_t)3 * CD * CD);
  _Float16* Wv16  = allocH((size_t)3 * CD * CD);
  _Float16* Wfu16 = allocH((size_t)CD * 384);
  // f16 activations, pixel-major [b][pix][C] unless noted
  _Float16* SD16   = allocH((size_t)BB * N0 * CD);
  _Float16* SS16   = allocH((size_t)BB * N1 * CD);
  _Float16* rec16  = allocH((size_t)BB * N1 * 32);
  _Float16* skipU16 = allocH((size_t)BB * N1 * CD);
  _Float16* xu16   = allocH((size_t)BB * N1 * CIN);
  _Float16* mrgd16 = allocH((size_t)BB * N1 * CD);
  _Float16* sgUp16 = allocH((size_t)BB * N1 * CD);
  _Float16* qB16   = allocH((size_t)3 * BB * CD * N1);  // channel-major
  _Float16* kB16   = allocH((size_t)3 * BB * CD * N1);  // channel-major
  _Float16* vT16   = allocH((size_t)3 * BB * N1 * CD);  // pixel-major
  _Float16* attnP16 = allocH((size_t)3 * BB * NHEAD * 1024);
  _Float16* triT16 = allocH((size_t)BB * N1 * 384);     // pixel-major
  _Float16* attO16 = allocH((size_t)BB * N1 * CD);
  _Float16* r1a16  = allocH((size_t)BB * N1 * CD);

  const dim3 blk256(256), blk128(128);
  auto ew = [](size_t tot) { return dim3((unsigned)((tot + 255) / 256)); };

  // --- weight prep ---
  wprep_conv_kernel<<<ew((size_t)9 * 32 * CD), blk256, 0, stream>>>(Wll, Wll16, 32, CD);
  wprep_conv_kernel<<<ew((size_t)9 * 96 * CD), blk256, 0, stream>>>(Whi, Whi16, 96, CD);
  wprep_conv_kernel<<<ew((size_t)9 * CD * CIN), blk256, 0, stream>>>(Wup, Wup16, CD, CIN);
  wprep_conv_kernel<<<ew((size_t)9 * CD * CD), blk256, 0, stream>>>(Wr1, Wr116, CD, CD);
  wprep_conv_kernel<<<ew((size_t)9 * CD * CD), blk256, 0, stream>>>(Wr2, Wr216, CD, CD);
  wcvt_kernel<<<ew((size_t)CD * 32), blk256, 0, stream>>>(Wadj, Wadj16, (size_t)CD * 32);
  wcvt_kernel<<<ew((size_t)3 * CD * CD), blk256, 0, stream>>>(Wq, Wq16, (size_t)3 * CD * CD);
  wcvt_kernel<<<ew((size_t)3 * CD * CD), blk256, 0, stream>>>(Wk, Wk16, (size_t)3 * CD * CD);
  wcvt_kernel<<<ew((size_t)3 * CD * CD), blk256, 0, stream>>>(Wv, Wv16, (size_t)3 * CD * CD);
  wcvt_kernel<<<ew((size_t)CD * 384), blk256, 0, stream>>>(Wfu, Wfu16, (size_t)CD * 384);

  // --- f16 pixel-major twins of raw inputs ---
  cvt_pm_kernel<<<ew((size_t)BB * CD * (N0 / 4)), blk256, 0, stream>>>(SD, SD16, BB, CD, N0);
  cvt_pm_kernel<<<ew((size_t)BB * CD * (N1 / 4)), blk256, 0, stream>>>(SS, SS16, BB, CD, N1);

  // --- detail branch: conv(ll)+conv(high) into one 128ch buffer, GN+tanh ---
  conv3x3_wmma_kernel<<<dim3(2, N0 / 64, BB), blk128, 0, stream>>>(
      SD16, Wll16, bufA, CD, 32, H0, W0C, 128, 0);
  conv3x3_wmma_kernel<<<dim3(6, N0 / 64, BB), blk128, 0, stream>>>(
      SD16, Whi16, bufA, CD, 96, H0, W0C, 128, 32);
  gn_stats_kernel<<<BB * 8, blk256, 0, stream>>>(bufA, nullptr, stats, 128, 0, 32, 8, N0);
  gn_apply_kernel<<<ew((size_t)BB * 32 * (N0 / 4)), blk256, 0, stream>>>(
      bufA, nullptr, bufA, nullptr, gllg, gllb, stats, wscl, 2, BB, 128, 0, 32, 8, N0);
  gn_stats_kernel<<<BB * 8, blk256, 0, stream>>>(bufA, nullptr, stats, 128, 32, 96, 8, N0);
  gn_apply_kernel<<<ew((size_t)BB * 96 * (N0 / 4)), blk256, 0, stream>>>(
      bufA, nullptr, bufA, nullptr, ghig, ghib, stats, wscl, 2, BB, 128, 32, 96, 8, N0);
  idwt_kernel<<<ew((size_t)BB * 32 * N0), blk256, 0, stream>>>(bufA, rec16);

  // --- skip_up = relu(GN(W_adj @ rec)) ---
  gemm1x1_wmma_kernel<<<dim3(CD / 16, N1 / 64, BB), blk128, 0, stream>>>(
      rec16, Wadj16, nullptr, skipU, nullptr, nullptr, 32, CD, N1);
  gn_stats_kernel<<<BB * 8, blk256, 0, stream>>>(skipU, nullptr, stats, CD, 0, CD, 8, N1);
  gn_apply_kernel<<<ew((size_t)BB * CD * (N1 / 4)), blk256, 0, stream>>>(
      skipU, nullptr, skipU, skipU16, gadjg, gadjb, stats, nullptr, 1, BB, CD, 0, CD, 8, N1);

  // --- merged = relu(BN(conv(resize(x)))) + skip_up ---
  resize_pm_kernel<<<ew((size_t)BB * CIN * N1), blk256, 0, stream>>>(
      X, xu16, BB, CIN, H0, W0C, H1, W1C);
  conv3x3_wmma_kernel<<<dim3(CD / 16, N1 / 64, BB), blk128, 0, stream>>>(
      xu16, Wup16, mrgd, CIN, CD, H1, W1C, CD, 0);
  bn_stats_kernel<<<CD, blk256, 0, stream>>>(mrgd, stats, BB, CD, N1);
  bn_apply_kernel<<<ew((size_t)BB * CD * (N1 / 4)), blk256, 0, stream>>>(
      mrgd, skipU, mrgd, mrgd16, bnupg, bnupb, stats, BB, CD, N1);

  // --- resize skip_global (skip_structure already at target size) ---
  resize_pm_kernel<<<ew((size_t)BB * CD * N1), blk256, 0, stream>>>(
      SG, sgUp16, BB, CD, H0, W0C, H1, W1C);

  // --- q/k/v projections ---
  for (int p = 0; p < 3; ++p) {
    const _Float16* feat = (p == 0) ? skipU16 : (p == 1) ? SS16 : sgUp16;
    const size_t po      = (size_t)p * BB * CD * N1;  // == p*BB*N1*CD
    gemm1x1_wmma_kernel<<<dim3(CD / 16, N1 / 64, BB), blk128, 0, stream>>>(
        mrgd16, Wq16 + (size_t)p * CD * CD, bq + p * CD, nullptr, qB16 + po,
        nullptr, CD, CD, N1);
    gemm1x1_wmma_kernel<<<dim3(CD / 16, N1 / 64, BB), blk128, 0, stream>>>(
        feat, Wk16 + (size_t)p * CD * CD, bk + p * CD, nullptr, kB16 + po,
        nullptr, CD, CD, N1);
    gemm1x1_wmma_kernel<<<dim3(CD / 16, N1 / 64, BB), blk128, 0, stream>>>(
        feat, Wv16 + (size_t)p * CD * CD, bv + p * CD, nullptr, nullptr,
        vT16 + po, CD, CD, N1);
  }

  // --- attention ---
  attn_scores_kernel<<<3 * BB * NHEAD, blk128, 0, stream>>>(qB16, kB16, attnP16, N1);
  attn_apply_kernel<<<dim3(N1 / 64, 3 * BB * NHEAD), blk128, 0, stream>>>(
      attnP16, vT16, triT16, N1);

  // --- fused = silu(GN(W_fuse @ tri + b_fuse)) ---
  gemm1x1_wmma_kernel<<<dim3(CD / 16, N1 / 64, BB), blk128, 0, stream>>>(
      triT16, Wfu16, bfu, fused, nullptr, nullptr, 384, CD, N1);
  gn_stats_kernel<<<BB * 8, blk256, 0, stream>>>(fused, nullptr, stats, CD, 0, CD, 8, N1);
  gn_apply_kernel<<<ew((size_t)BB * CD * (N1 / 4)), blk256, 0, stream>>>(
      fused, nullptr, fused, nullptr, gfg, gfb, stats, nullptr, 3, BB, CD, 0, CD, 8, N1);

  // --- att_out = GN(merged + fused) -> f16 pixel-major (feeds conv r1) ---
  gn_stats_kernel<<<BB * 8, blk256, 0, stream>>>(mrgd, fused, stats, CD, 0, CD, 8, N1);
  gn_apply_kernel<<<ew((size_t)BB * CD * (N1 / 4)), blk256, 0, stream>>>(
      mrgd, fused, nullptr, attO16, gog, gob, stats, nullptr, 0, BB, CD, 0, CD, 8, N1);

  // --- residual tail ---
  conv3x3_wmma_kernel<<<dim3(CD / 16, N1 / 64, BB), blk128, 0, stream>>>(
      attO16, Wr116, r1b, CD, CD, H1, W1C, CD, 0);
  bn_stats_kernel<<<CD, blk256, 0, stream>>>(r1b, stats, BB, CD, N1);
  bn_apply_kernel<<<ew((size_t)BB * CD * (N1 / 4)), blk256, 0, stream>>>(
      r1b, nullptr, nullptr, r1a16, bn1g, bn1b, stats, BB, CD, N1);

  conv3x3_wmma_kernel<<<dim3(CD / 16, N1 / 64, BB), blk128, 0, stream>>>(
      r1a16, Wr216, fused, CD, CD, H1, W1C, CD, 0);
  bn_stats_kernel<<<CD, blk256, 0, stream>>>(fused, stats, BB, CD, N1);
  bn_apply_kernel<<<ew((size_t)BB * CD * (N1 / 4)), blk256, 0, stream>>>(
      fused, nullptr, (float*)d_out, nullptr, bn2g, bn2b, stats, BB, CD, N1);
}
